// GraphAttentionLayer_83537113907600
// MI455X (gfx1250) — compile-verified, matrix operand order checked by reading, not run on hardware
//
#include <hip/hip_runtime.h>
#include <hip/hip_bf16.h>
#include <stdint.h>

#define NROWS 8192
#define F_IN  256
#define F_OUT 128
#define ALPHA 0.2f
#define LOG2E 1.4426950408889634f
#define JSPLIT 4
#define KT 32
#define NKT ((NROWS / JSPLIT) / KT)      // 64 k-tiles per block
#define TILE_BYTES 8192                  // 8 ntiles * 32 lanes * 16 elems * 2B

typedef __attribute__((ext_vector_type(16))) __bf16 v16bf;
typedef __attribute__((ext_vector_type(8)))  float  v8f;

// ---------------- K1: Wh = x @ W^T (tiny GEMM, 0.54 GFLOP; plain FMA is fine) ---------
__global__ __launch_bounds__(256) void k1_gemm_wh(const float* __restrict__ x,
                                                  const float* __restrict__ W,
                                                  float* __restrict__ Wh) {
  __shared__ float xs[32][F_IN + 1];
  const int rbase = blockIdx.x * 32;
  for (int idx = threadIdx.x; idx < 32 * (F_IN / 4); idx += 256) {
    int r = idx >> 6, c4 = idx & 63;
    float4 v = *(const float4*)(x + (size_t)(rbase + r) * F_IN + c4 * 4);
    xs[r][c4 * 4 + 0] = v.x; xs[r][c4 * 4 + 1] = v.y;
    xs[r][c4 * 4 + 2] = v.z; xs[r][c4 * 4 + 3] = v.w;
  }
  __syncthreads();
  const int col = threadIdx.x & 127;
  const int rg  = threadIdx.x >> 7;
  float acc[16];
#pragma unroll
  for (int r = 0; r < 16; ++r) acc[r] = 0.f;
  const float* wrow = W + (size_t)col * F_IN;
  for (int k = 0; k < F_IN; k += 4) {
    float4 wv = *(const float4*)(wrow + k);
#pragma unroll
    for (int r = 0; r < 16; ++r) {
      const float* xr = &xs[rg * 16 + r][k];
      acc[r] += xr[0] * wv.x + xr[1] * wv.y + xr[2] * wv.z + xr[3] * wv.w;
    }
  }
#pragma unroll
  for (int r = 0; r < 16; ++r)
    Wh[(size_t)(rbase + rg * 16 + r) * F_OUT + col] = acc[r];
}

// ---------------- K2: s1/s2 row scores + pack Wh into bf16 WMMA-B lane layout ---------
__device__ inline unsigned short f32_to_bf16_rne(float f) {
  unsigned u = __builtin_bit_cast(unsigned, f);
  unsigned r = u + 0x7FFFu + ((u >> 16) & 1u);
  return (unsigned short)(r >> 16);
}

__global__ __launch_bounds__(256) void k2_rowstats(const float* __restrict__ Wh,
                                                   const float* __restrict__ a,
                                                   float* __restrict__ s1,
                                                   float* __restrict__ s2,
                                                   unsigned short* __restrict__ whb) {
  const int i = blockIdx.x * 256 + threadIdx.x;       // row index j of Wh (= K of B matrix)
  const float* row = Wh + (size_t)i * F_OUT;
  const int ktile = i >> 5;
  const int kin   = i & 31;
  const int hi    = (kin >> 3) & 1;                   // lanes 16..31 hold K%16 >= 8
  const int e     = (kin & 7) | ((kin & 16) >> 1);    // slot 0..15 within v16bf operand
  float acc1 = 0.f, acc2 = 0.f;
#pragma unroll 4
  for (int n = 0; n < F_OUT; ++n) {
    float v = row[n];
    acc1 += v * a[n];
    acc2 += v * a[F_OUT + n];
    int lane = (n & 15) + (hi << 4);
    int nt   = n >> 4;
    whb[(((size_t)ktile * 8 + nt) * 32 + lane) * 16 + e] = f32_to_bf16_rne(v);
  }
  s1[i] = acc1;
  s2[i] = acc2;
}

// ---------------- K2b: global max of s2 (gives a valid fixed softmax shift) -----------
__global__ __launch_bounds__(256) void k2b_s2max(const float* __restrict__ s2,
                                                 float* __restrict__ s2max) {
  __shared__ float sm[256];
  float m = -3.0e38f;
  for (int j = threadIdx.x; j < NROWS; j += 256) m = fmaxf(m, s2[j]);
  sm[threadIdx.x] = m;
  __syncthreads();
  for (int w = 128; w > 0; w >>= 1) {
    if (threadIdx.x < w) sm[threadIdx.x] = fmaxf(sm[threadIdx.x], sm[threadIdx.x + w]);
    __syncthreads();
  }
  if (threadIdx.x == 0) s2max[0] = sm[0];
}

// ---------------- CDNA5 async-to-LDS helpers ------------------------------------------
__device__ inline void async_b128(unsigned lds_off, const void* gsrc) {
  asm volatile("global_load_async_to_lds_b128 %0, %1, off"
               :: "v"(lds_off), "v"((unsigned long long)(uintptr_t)gsrc)
               : "memory");
}
__device__ inline void wait_async_le2() {
#if defined(__has_builtin) && __has_builtin(__builtin_amdgcn_s_wait_asynccnt)
  __builtin_amdgcn_s_wait_asynccnt(2);
#else
  asm volatile("s_wait_asynccnt 0x2" ::: "memory");
#endif
}
__device__ inline void wait_async_le0() {
#if defined(__has_builtin) && __has_builtin(__builtin_amdgcn_s_wait_asynccnt)
  __builtin_amdgcn_s_wait_asynccnt(0);
#else
  asm volatile("s_wait_asynccnt 0x0" ::: "memory");
#endif
}

// ---------------- K3: fused mask + leaky-relu + exp + (P @ Wh) flash pass -------------
// grid (64 row-blocks, JSPLIT j-splits), 256 threads = 8 waves, 16 rows per wave.
__global__ __launch_bounds__(256) void k3_attn(const int* __restrict__ adj,
                                               const float* __restrict__ s1,
                                               const float* __restrict__ s2,
                                               const float* __restrict__ s2max,
                                               const unsigned short* __restrict__ whb,
                                               float* __restrict__ Cpart,
                                               float* __restrict__ lpart) {
  __shared__ unsigned char bstage[2][TILE_BYTES];
  const int tid  = threadIdx.x;
  const int wave = tid >> 5;
  const int lane = tid & 31;
  const int hi   = lane >> 4;
  const int rb   = blockIdx.x;
  const int s    = blockIdx.y;
  const int row  = rb * 128 + wave * 16 + (lane & 15);   // A-matrix: M = lane%16

  const float s1v   = s1[row];
  const float mxb   = s1v + s2max[0];
  const float shift = mxb > 0.f ? mxb : ALPHA * mxb;     // >= row max of e (lrelu monotone)

  const int kt0 = (s * (NROWS / JSPLIT)) >> 5;

  { // stage first B tile (bf16 Wh slice, already in WMMA-B lane layout)
    const unsigned char* g = (const unsigned char*)whb + (size_t)kt0 * TILE_BYTES + tid * 32;
    unsigned l = (unsigned)(uintptr_t)&bstage[0][tid * 32];
    async_b128(l, g);
    async_b128(l + 16, g + 16);
  }

  v8f acc[8];
#pragma unroll
  for (int n = 0; n < 8; ++n) acc[n] = v8f{0.f, 0.f, 0.f, 0.f, 0.f, 0.f, 0.f, 0.f};
  float lacc = 0.f;

  const int kbase = hi ? 8 : 0;   // A lane layout: lanes 0-15 K∈{0..7,16..23}; 16-31 +8
  int buf = 0;
  for (int t = 0; t < NKT; ++t) {
    const int akt = kt0 + t;
    if (t + 1 < NKT) {
      const unsigned char* g = (const unsigned char*)whb + (size_t)(akt + 1) * TILE_BYTES + tid * 32;
      unsigned l = (unsigned)(uintptr_t)&bstage[buf ^ 1][tid * 32];
      async_b128(l, g);
      async_b128(l + 16, g + 16);
      wait_async_le2();            // loads retire in order -> current buffer is complete
    } else {
      wait_async_le0();
    }
    __syncthreads();               // bstage[buf] visible to all 8 waves

    const int jb = akt * KT;
    const int*   arow = adj + (size_t)row * NROWS + jb + kbase;
    const float* srow = s2 + jb + kbase;
    __builtin_prefetch(arow + 64, 0, 3);   // global_prefetch_b8: next adj line of this row

    int4 ai0 = *(const int4*)(arow + 0);
    int4 ai1 = *(const int4*)(arow + 4);
    int4 ai2 = *(const int4*)(arow + 16);
    int4 ai3 = *(const int4*)(arow + 20);
    float4 sf0 = *(const float4*)(srow + 0);
    float4 sf1 = *(const float4*)(srow + 4);
    float4 sf2 = *(const float4*)(srow + 16);
    float4 sf3 = *(const float4*)(srow + 20);

    const int av[16] = {ai0.x, ai0.y, ai0.z, ai0.w, ai1.x, ai1.y, ai1.z, ai1.w,
                        ai2.x, ai2.y, ai2.z, ai2.w, ai3.x, ai3.y, ai3.z, ai3.w};
    const float sv[16] = {sf0.x, sf0.y, sf0.z, sf0.w, sf1.x, sf1.y, sf1.z, sf1.w,
                          sf2.x, sf2.y, sf2.z, sf2.w, sf3.x, sf3.y, sf3.z, sf3.w};

    v16bf A;
#pragma unroll
    for (int e = 0; e < 16; ++e) {
      float ev = s1v + sv[e];
      ev = ev > 0.f ? ev : ALPHA * ev;                      // leaky_relu
      float p = (av[e] > 0) ? __builtin_amdgcn_exp2f((ev - shift) * LOG2E) : 0.f;
      lacc += p;                                            // softmax denominator partial
      A[e] = (__bf16)p;
    }

    const unsigned char* lbase = &bstage[buf][0];
#pragma unroll
    for (int nt = 0; nt < 8; ++nt) {
      union { uint4 q[2]; v16bf v; } B;
      const uint4* lp = (const uint4*)(lbase + ((size_t)nt * 32 + lane) * 32);
      B.q[0] = lp[0];
      B.q[1] = lp[1];
      acc[nt] = __builtin_amdgcn_wmma_f32_16x16x32_bf16(false, A, false, B.v,
                                                        (short)0, acc[nt], false, false);
    }
    __syncthreads();               // all waves done with bstage[buf] before reuse
    buf ^= 1;
  }

  // lane and lane^16 hold disjoint K subsets of the same row -> combine
  float lsum = lacc + __shfl_xor(lacc, 16, 32);

  const size_t pbase = ((size_t)rb * JSPLIT + s) * 128;
  if (hi == 0) lpart[pbase + wave * 16 + lane] = lsum;

  // C/D layout: VGPR r -> (M = r + 8*hi, N = nt*16 + lane%16)
#pragma unroll
  for (int nt = 0; nt < 8; ++nt) {
#pragma unroll
    for (int r = 0; r < 8; ++r) {
      int lr = wave * 16 + r + hi * 8;
      Cpart[(pbase + lr) * F_OUT + nt * 16 + (lane & 15)] = acc[nt][r];
    }
  }
}

// ---------------- K4: combine j-splits, divide by denominator, ELU --------------------
__global__ __launch_bounds__(256) void k4_combine(const float* __restrict__ Cpart,
                                                  const float* __restrict__ lpart,
                                                  float* __restrict__ out) {
  const int idx = blockIdx.x * 256 + threadIdx.x;
  const int row = idx >> 7;
  const int col = idx & 127;
  const int rbk = row >> 7;
  const int lr  = row & 127;
  float c = 0.f, l = 0.f;
#pragma unroll
  for (int s = 0; s < JSPLIT; ++s) {
    size_t pb = ((size_t)rbk * JSPLIT + s) * 128;
    c += Cpart[(pb + lr) * F_OUT + col];
    l += lpart[pb + lr];
  }
  float h = c / l;
  out[idx] = h > 0.f ? h : (__builtin_amdgcn_exp2f(h * LOG2E) - 1.f);
}

// ---------------- host ----------------------------------------------------------------
extern "C" void kernel_launch(void* const* d_in, const int* in_sizes, int n_in,
                              void* d_out, int out_size, void* d_ws, size_t ws_size,
                              hipStream_t stream) {
  (void)in_sizes; (void)n_in; (void)out_size; (void)ws_size;
  const float* x   = (const float*)d_in[0];
  const int*   adj = (const int*)d_in[1];
  const float* W   = (const float*)d_in[2];
  const float* a   = (const float*)d_in[3];
  float* out = (float*)d_out;

  char* ws = (char*)d_ws;
  size_t off = 0;
  auto alloc = [&](size_t bytes) -> void* {
    void* p = ws + off;
    off = (off + bytes + 255) & ~(size_t)255;
    return p;
  };
  float*          Wh    = (float*)alloc((size_t)NROWS * F_OUT * 4);          // 4 MB
  unsigned short* whb   = (unsigned short*)alloc((size_t)NROWS * F_OUT * 2); // 2 MB
  float*          s1    = (float*)alloc((size_t)NROWS * 4);
  float*          s2    = (float*)alloc((size_t)NROWS * 4);
  float*          s2m   = (float*)alloc(256);
  float*          Cpart = (float*)alloc((size_t)64 * JSPLIT * 128 * F_OUT * 4); // 16 MB
  float*          lpart = (float*)alloc((size_t)64 * JSPLIT * 128 * 4);

  k1_gemm_wh<<<NROWS / 32, 256, 0, stream>>>(x, W, Wh);
  k2_rowstats<<<NROWS / 256, 256, 0, stream>>>(Wh, a, s1, s2, whb);
  k2b_s2max<<<1, 256, 0, stream>>>(s2, s2m);
  k3_attn<<<dim3(64, JSPLIT), 256, 0, stream>>>(adj, s1, s2, s2m, whb, Cpart, lpart);
  k4_combine<<<(NROWS * F_OUT) / 256, 256, 0, stream>>>(Cpart, lpart, out);
}
